// MoE_21182778704758
// MI455X (gfx1250) — matrix operand order repaired and checked by reading, not compiled, not run on hardware
//
#include <hip/hip_runtime.h>
#include <hip/hip_bf16.h>
#include <math.h>
#include <stdint.h>

// ---------------- problem constants ----------------
#define D_MODEL 768
#define D_FF    3072
#define N_EXP   8
#define T_TOK   4096      // B*S = 2*2048
#define TMAX    4096      // max tokens per expert bucket

// workspace layout (bytes)
#define OFF_CNT   0            // int[8]
#define OFF_BASE  32           // int[8]
#define OFF_TOK   64           // int[8*4096]   = 128 KB
#define OFF_WGT   (64 + 8*TMAX*4)              // float[8*4096] = 128 KB
#define OFF_H     (OFF_WGT + 8*TMAX*4)         // bf16[8192 * 3072] ~= 50 MB

typedef __attribute__((ext_vector_type(16))) __bf16 v16bf;
typedef __attribute__((ext_vector_type(8)))  __bf16 v8bf;
typedef __attribute__((ext_vector_type(8)))  float  v8f;

// ---------------- kernel 0: zero output + counters ----------------
__global__ __launch_bounds__(256) void k_zero(float* __restrict__ out, int n,
                                              int* __restrict__ cnt) {
    int i = blockIdx.x * blockDim.x + threadIdx.x;
    int stride = gridDim.x * blockDim.x;
    for (; i < n; i += stride) out[i] = 0.0f;
    if (blockIdx.x == 0 && threadIdx.x < N_EXP) cnt[threadIdx.x] = 0;
}

// ---------------- kernel 1: gating (logits, top-2, softmax, bucketize) ----
__global__ __launch_bounds__(256) void k_gate(const float* __restrict__ x,
                                              const float* __restrict__ Wr,
                                              int* __restrict__ cnt,
                                              int* __restrict__ tok,
                                              float* __restrict__ wgt) {
    int t = blockIdx.x * blockDim.x + threadIdx.x;
    if (t >= T_TOK) return;
    const float* xr = x + (size_t)t * D_MODEL;
    float acc[N_EXP];
#pragma unroll
    for (int e = 0; e < N_EXP; ++e) acc[e] = 0.0f;
    for (int d = 0; d < D_MODEL; d += 4) {
        float4 xv = *(const float4*)(xr + d);
        const float* wr = Wr + (size_t)d * N_EXP;
#pragma unroll
        for (int e = 0; e < N_EXP; ++e)
            acc[e] += xv.x * wr[e] + xv.y * wr[e + 8] +
                      xv.z * wr[e + 16] + xv.w * wr[e + 24];
    }
    int i0 = 0;
#pragma unroll
    for (int e = 1; e < N_EXP; ++e) if (acc[e] > acc[i0]) i0 = e;
    int i1 = (i0 == 0) ? 1 : 0;
#pragma unroll
    for (int e = 0; e < N_EXP; ++e)
        if (e != i0 && acc[e] > acc[i1]) i1 = e;
    float e1 = __expf(acc[i1] - acc[i0]);          // l0 >= l1
    float inv = 1.0f / (1.0f + e1);

    int p0 = atomicAdd(&cnt[i0], 1);
    tok[i0 * TMAX + p0] = t; wgt[i0 * TMAX + p0] = inv;
    int p1 = atomicAdd(&cnt[i1], 1);
    tok[i1 * TMAX + p1] = t; wgt[i1 * TMAX + p1] = e1 * inv;
}

// ---------------- kernel 2: exclusive scan of counts -----------------
__global__ void k_scan(const int* __restrict__ cnt, int* __restrict__ base) {
    if (threadIdx.x == 0 && blockIdx.x == 0) {
        int s = 0;
#pragma unroll
        for (int e = 0; e < N_EXP; ++e) { base[e] = s; s += cnt[e]; }
    }
}

// ---- fragment-layout helpers (ISA 16-bit A 16x32 / B 32x16, wave32) ----
__device__ __forceinline__ int a_chunk_off(int m, int c) {
    return ((m >> 4) << 9) + (((m & 15) + ((c >> 1) << 4)) << 4) + ((c & 1) << 3);
}
__device__ __forceinline__ int b_off(int n, int kg) {
    return ((n >> 4) << 9) + (((n & 15) + (kg << 4)) << 4);
}
__device__ __forceinline__ v8bf cvt8(float4 f0, float4 f1) {
    v8bf p;
    p[0] = (__bf16)f0.x; p[1] = (__bf16)f0.y; p[2] = (__bf16)f0.z; p[3] = (__bf16)f0.w;
    p[4] = (__bf16)f1.x; p[5] = (__bf16)f1.y; p[6] = (__bf16)f1.z; p[7] = (__bf16)f1.w;
    return p;
}

// ---------------- kernel 3: grouped GEMM1  H = relu(Xg @ W1[e] + b1[e]) ----
// CTA tile 128x128, K-step 32, 8 waves, double-buffered LDS, 1 barrier/iter.
__global__ __launch_bounds__(256) void k_ffn1(const float* __restrict__ x,
                                              const float* __restrict__ W1,
                                              const float* __restrict__ b1,
                                              const int* __restrict__ cnt,
                                              const int* __restrict__ base,
                                              const int* __restrict__ tok,
                                              __bf16* __restrict__ H) {
    const int e = blockIdx.z;
    const int count = cnt[e];
    const int m0 = blockIdx.x * 128;
    if (m0 >= count) return;
    const int n0 = blockIdx.y * 128;

    __shared__ __align__(32) __bf16 sA[2][8 * 512];
    __shared__ __align__(32) __bf16 sB[2][8 * 512];
    __shared__ int sTok[128];

    const int tid = threadIdx.x;
    const int lane = tid & 31, w = tid >> 5;
    const int wm = w >> 1, wn = w & 1;

    if (tid < 128) {
        int r = m0 + tid;
        sTok[tid] = tok[e * TMAX + ((r < count) ? r : (count - 1))];
    }
    __syncthreads();

    // staging roles (fixed per thread)
    const int am = tid >> 2, ac = tid & 3;
    const float* arow0 = x + (size_t)sTok[am] * D_MODEL + ac * 8;
    const float* arow1 = x + (size_t)sTok[am + 64] * D_MODEL + ac * 8;
    const int adoff0 = a_chunk_off(am, ac);
    const int adoff1 = a_chunk_off(am + 64, ac);
    const int bn = tid & 127, bkg = tid >> 7;
    const float* bcol = W1 + (size_t)e * D_MODEL * D_FF + n0 + bn +
                        (size_t)(bkg * 16) * D_FF;
    const int bdoff = b_off(bn, bkg);

    float4 ra0, ra1, ra2, ra3;
    float fb[16];
    auto load_regs = [&](int k0) {
        ra0 = *(const float4*)(arow0 + k0);
        ra1 = *(const float4*)(arow0 + k0 + 4);
        ra2 = *(const float4*)(arow1 + k0);
        ra3 = *(const float4*)(arow1 + k0 + 4);
        const float* bp = bcol + (size_t)k0 * D_FF;
#pragma unroll
        for (int kk = 0; kk < 16; ++kk) fb[kk] = bp[(size_t)kk * D_FF];
    };
    auto store_lds = [&](int sel) {
        *(v8bf*)(&sA[sel][adoff0]) = cvt8(ra0, ra1);
        *(v8bf*)(&sA[sel][adoff1]) = cvt8(ra2, ra3);
        *(v8bf*)(&sB[sel][bdoff]) = cvt8(make_float4(fb[0], fb[1], fb[2], fb[3]),
                                         make_float4(fb[4], fb[5], fb[6], fb[7]));
        *(v8bf*)(&sB[sel][bdoff + 8]) =
            cvt8(make_float4(fb[8], fb[9], fb[10], fb[11]),
                 make_float4(fb[12], fb[13], fb[14], fb[15]));
    };

    v8f acc[2][4] = {};
    const int NIT = D_MODEL / 32;

    load_regs(0);
    store_lds(0);
    __syncthreads();

    for (int it = 0; it < NIT; ++it) {
        const int sel = it & 1;
        if (it + 1 < NIT) load_regs((it + 1) * 32);   // overlap with WMMAs

        v16bf a0 = *(const v16bf*)(&sA[sel][((2 * wm) << 9) + lane * 16]);
        v16bf a1 = *(const v16bf*)(&sA[sel][((2 * wm + 1) << 9) + lane * 16]);
#pragma unroll
        for (int s = 0; s < 4; ++s) {
            v16bf b = *(const v16bf*)(&sB[sel][((wn * 4 + s) << 9) + lane * 16]);
            acc[0][s] = __builtin_amdgcn_wmma_f32_16x16x32_bf16(
                false, a0, false, b, (short)0, acc[0][s], false, false);
            acc[1][s] = __builtin_amdgcn_wmma_f32_16x16x32_bf16(
                false, a1, false, b, (short)0, acc[1][s], false, false);
        }
        if (it + 1 < NIT) store_lds(sel ^ 1);
        __syncthreads();
    }

    // epilogue: relu(+b1) -> bf16 H rows [base[e]+r]
    const int g = lane >> 4, nl = lane & 15;
    const int hbase = base[e];
    const bool full = (m0 + 128) <= count;
#pragma unroll
    for (int a = 0; a < 2; ++a) {
#pragma unroll
        for (int s = 0; s < 4; ++s) {
            int ncol = n0 + (wn * 4 + s) * 16 + nl;
            float bb = b1[e * D_FF + ncol];
            int rbase = m0 + (2 * wm + a) * 16 + 8 * g;
            __bf16* hp = H + (size_t)(hbase + rbase) * D_FF + ncol;
            if (full) {
#pragma unroll
                for (int v = 0; v < 8; ++v) {
                    float val = acc[a][s][v] + bb;
                    val = val > 0.0f ? val : 0.0f;
                    hp[(size_t)v * D_FF] = (__bf16)val;
                }
            } else {
#pragma unroll
                for (int v = 0; v < 8; ++v) {
                    if (rbase + v < count) {
                        float val = acc[a][s][v] + bb;
                        val = val > 0.0f ? val : 0.0f;
                        hp[(size_t)v * D_FF] = (__bf16)val;
                    }
                }
            }
        }
    }
}

// ---------------- kernel 4: grouped GEMM2  out += w * (H @ W2[e] + b2[e]) ---
__global__ __launch_bounds__(256) void k_ffn2(const __bf16* __restrict__ H,
                                              const float* __restrict__ W2,
                                              const float* __restrict__ b2,
                                              const int* __restrict__ cnt,
                                              const int* __restrict__ base,
                                              const int* __restrict__ tok,
                                              const float* __restrict__ wgt,
                                              float* __restrict__ out) {
    const int e = blockIdx.z;
    const int count = cnt[e];
    const int m0 = blockIdx.x * 128;
    if (m0 >= count) return;
    const int n0 = blockIdx.y * 128;

    __shared__ __align__(32) __bf16 sA[2][8 * 512];
    __shared__ __align__(32) __bf16 sB[2][8 * 512];
    __shared__ int   sTok[128];
    __shared__ float sW[128];

    const int tid = threadIdx.x;
    const int lane = tid & 31, w = tid >> 5;
    const int wm = w >> 1, wn = w & 1;
    const int hbase = base[e];

    if (tid < 128) {
        int r = m0 + tid;
        int rc = (r < count) ? r : (count - 1);
        sTok[tid] = tok[e * TMAX + rc];
        sW[tid]   = wgt[e * TMAX + rc];
    }

    // A rows in H are bf16 & contiguous: async global->LDS B128 copies.
    const int am = tid >> 2, ac = tid & 3;
    int r0 = m0 + am;       r0 = (r0 < count) ? r0 : (count - 1);
    int r1 = m0 + am + 64;  r1 = (r1 < count) ? r1 : (count - 1);
    const __bf16* asrc0 = H + (size_t)(hbase + r0) * D_FF + ac * 8;
    const __bf16* asrc1 = H + (size_t)(hbase + r1) * D_FF + ac * 8;
    const unsigned aoff0 = (unsigned)(uintptr_t)(&sA[0][a_chunk_off(am, ac)]);
    const unsigned aoff1 = (unsigned)(uintptr_t)(&sA[0][a_chunk_off(am + 64, ac)]);
    const unsigned abufstride = (unsigned)(uintptr_t)(&sA[1][0]) -
                                (unsigned)(uintptr_t)(&sA[0][0]);

    const int bn = tid & 127, bkg = tid >> 7;
    const float* bcol = W2 + (size_t)e * D_FF * D_MODEL + n0 + bn +
                        (size_t)(bkg * 16) * D_MODEL;
    const int bdoff = b_off(bn, bkg);

    float fb[16];
    auto issue_async_A = [&](int k0, int sel) {
        unsigned o0 = aoff0 + sel * abufstride;
        unsigned o1 = aoff1 + sel * abufstride;
        asm volatile("global_load_async_to_lds_b128 %0, %1, off"
                     :: "v"(o0), "v"(asrc0 + k0) : "memory");
        asm volatile("global_load_async_to_lds_b128 %0, %1, off"
                     :: "v"(o1), "v"(asrc1 + k0) : "memory");
    };
    auto load_B_regs = [&](int k0) {
        const float* bp = bcol + (size_t)k0 * D_MODEL;
#pragma unroll
        for (int kk = 0; kk < 16; ++kk) fb[kk] = bp[(size_t)kk * D_MODEL];
    };
    auto store_B_lds = [&](int sel) {
        *(v8bf*)(&sB[sel][bdoff]) = cvt8(make_float4(fb[0], fb[1], fb[2], fb[3]),
                                         make_float4(fb[4], fb[5], fb[6], fb[7]));
        *(v8bf*)(&sB[sel][bdoff + 8]) =
            cvt8(make_float4(fb[8], fb[9], fb[10], fb[11]),
                 make_float4(fb[12], fb[13], fb[14], fb[15]));
    };

    v8f acc[2][4] = {};
    const int NIT = D_FF / 32;

    issue_async_A(0, 0);
    load_B_regs(0);
    store_B_lds(0);
    asm volatile("s_wait_asynccnt 0x0" ::: "memory");
    __syncthreads();

    for (int it = 0; it < NIT; ++it) {
        const int sel = it & 1;
        if (it + 1 < NIT) {
            issue_async_A((it + 1) * 32, sel ^ 1);    // overlaps with WMMAs
            load_B_regs((it + 1) * 32);
        }

        v16bf a0 = *(const v16bf*)(&sA[sel][((2 * wm) << 9) + lane * 16]);
        v16bf a1 = *(const v16bf*)(&sA[sel][((2 * wm + 1) << 9) + lane * 16]);
#pragma unroll
        for (int s = 0; s < 4; ++s) {
            v16bf b = *(const v16bf*)(&sB[sel][((wn * 4 + s) << 9) + lane * 16]);
            acc[0][s] = __builtin_amdgcn_wmma_f32_16x16x32_bf16(
                false, a0, false, b, (short)0, acc[0][s], false, false);
            acc[1][s] = __builtin_amdgcn_wmma_f32_16x16x32_bf16(
                false, a1, false, b, (short)0, acc[1][s], false, false);
        }
        if (it + 1 < NIT) store_B_lds(sel ^ 1);
        asm volatile("s_wait_asynccnt 0x0" ::: "memory");
        __syncthreads();
    }

    // epilogue: exactly 2 commutative f32 atomic adds per out element
    const int g = lane >> 4, nl = lane & 15;
    const bool full = (m0 + 128) <= count;
#pragma unroll
    for (int a = 0; a < 2; ++a) {
#pragma unroll
        for (int s = 0; s < 4; ++s) {
            int ncol = n0 + (wn * 4 + s) * 16 + nl;
            float bb = b2[e * D_MODEL + ncol];
            int mb = (2 * wm + a) * 16 + 8 * g;
            if (full) {
#pragma unroll
                for (int v = 0; v < 8; ++v) {
                    int m = mb + v;
                    atomicAdd(&out[(size_t)sTok[m] * D_MODEL + ncol],
                              sW[m] * (acc[a][s][v] + bb));
                }
            } else {
#pragma unroll
                for (int v = 0; v < 8; ++v) {
                    int m = mb + v;
                    if (m0 + m < count) {
                        atomicAdd(&out[(size_t)sTok[m] * D_MODEL + ncol],
                                  sW[m] * (acc[a][s][v] + bb));
                    }
                }
            }
        }
    }
}

// ---------------- host launch ----------------
extern "C" void kernel_launch(void* const* d_in, const int* in_sizes, int n_in,
                              void* d_out, int out_size, void* d_ws, size_t ws_size,
                              hipStream_t stream) {
    const float* x  = (const float*)d_in[0];   // [T, 768]
    const float* Wr = (const float*)d_in[1];   // [768, 8]
    const float* W1 = (const float*)d_in[2];   // [8, 768, 3072]
    const float* b1 = (const float*)d_in[3];   // [8, 3072]
    const float* W2 = (const float*)d_in[4];   // [8, 3072, 768]
    const float* b2 = (const float*)d_in[5];   // [8, 768]
    float* out = (float*)d_out;                // [T, 768]

    char* ws = (char*)d_ws;
    int*    cnt  = (int*)(ws + OFF_CNT);
    int*    base = (int*)(ws + OFF_BASE);
    int*    tok  = (int*)(ws + OFF_TOK);
    float*  wgt  = (float*)(ws + OFF_WGT);
    __bf16* H    = (__bf16*)(ws + OFF_H);

    k_zero<<<1024, 256, 0, stream>>>(out, T_TOK * D_MODEL, cnt);
    k_gate<<<T_TOK / 256, 256, 0, stream>>>(x, Wr, cnt, tok, wgt);
    k_scan<<<1, 32, 0, stream>>>(cnt, base);
    k_ffn1<<<dim3(TMAX / 128, D_FF / 128, N_EXP), 256, 0, stream>>>(
        x, W1, b1, cnt, base, tok, H);
    k_ffn2<<<dim3(TMAX / 128, D_MODEL / 128, N_EXP), 256, 0, stream>>>(
        H, W2, b2, cnt, base, tok, wgt, out);
}
